// AgentHetGNN_6519760355606
// MI455X (gfx1250) — compile-verified, matrix operand order checked by reading, not run on hardware
//
#include <hip/hip_runtime.h>
#include <hip/hip_bf16.h>

typedef __attribute__((ext_vector_type(8)))  __bf16 v8bf;
typedef __attribute__((ext_vector_type(16))) __bf16 v16bf;
typedef __attribute__((ext_vector_type(8)))  float  v8f;

#define HD   256
#define NTOT 24576
#define NPT  8192          // agents per type

// epilogue modes
#define EPI_STORE_F32     0
#define EPI_RELU_F32      1
#define EPI_ACCUM_F32     2
#define EPI_RELU_BF16     3
#define EPI_SILU_MUL_BF16 4
#define EPI_RESID_F32     5

// ---------------------------------------------------------------------------
// bf16 WMMA GEMM: C[M x Ncols] = epi(A[M x K] @ Bt^T + bias)
//   A  : bf16 row-major, leading dim lda
//   Bt : bf16 stored N-major ([Ncols][K]) so B tiles load exactly like A tiles
//   per-dst-type weights via strideBt/strideBias (type = row0 / 8192)
//
// Block tile 128x128, BK=32, 256 threads = 8 waves in 4x2.
// Wave tile 32x64 -> 2x4 = 8 x v_wmma_f32_16x16x32_bf16 per K-step.
// Double-buffered LDS + register prefetch of the next K-tile so the
// global_load -> s_wait_loadcnt lands after a full compute stage.
// ---------------------------------------------------------------------------
template<int EPI>
__launch_bounds__(256)
__global__ void gemm_bf16_wmma(const __bf16* __restrict__ A, int lda,
                               const __bf16* __restrict__ Bt, long strideBt, int K,
                               const float* __restrict__ bias, int strideBias,
                               float* __restrict__ Cf, __bf16* __restrict__ Cb, int ldc,
                               const float* __restrict__ aux1, int lda1,
                               const float* __restrict__ aux2, int lda2,
                               int Ncols)
{
    // rows padded to 40 bf16 (80B = 20 dwords) -> conflict-light b128 accesses
    __shared__ __align__(16) __bf16 As[2][128][40];
    __shared__ __align__(16) __bf16 Bs[2][128][40];

    const int tid  = threadIdx.x;
    const int row0 = blockIdx.y * 128;
    const int col0 = blockIdx.x * 128;
    const int typ  = row0 / NPT;                 // dst agent type of this row tile
    const __bf16* Bp    = Bt + (long)typ * strideBt;
    const float*  biasp = bias + (long)typ * strideBias;

    const int wid  = tid >> 5, lane = tid & 31;
    const int wm   = wid >> 1, wn   = wid & 1;   // 4 x 2 wave grid
    const int l16  = lane & 15, half = lane >> 4;

    v8f acc[2][4] = {};

    // global->LDS staging: each thread owns one 128-row tile row pair
    const int lr = tid >> 1;                     // 0..127 tile row
    const int lc = (tid & 1) * 16;               // 16-element half-row (2 x b128)

    const __bf16* ag = A  + (long)(row0 + lr) * lda + lc;
    const __bf16* bg = Bp + (long)(col0 + lr) * K   + lc;

    // prefetch tile 0 into registers
    v8bf rA0 = *(const v8bf*)(ag);
    v8bf rA1 = *(const v8bf*)(ag + 8);
    v8bf rB0 = *(const v8bf*)(bg);
    v8bf rB1 = *(const v8bf*)(bg + 8);

    const int nk = K >> 5;
    int buf = 0;
    for (int it = 0; it < nk; ++it) {
        // commit prefetched tile to LDS[buf]
        *(v8bf*)(&As[buf][lr][lc])     = rA0;
        *(v8bf*)(&As[buf][lr][lc + 8]) = rA1;
        *(v8bf*)(&Bs[buf][lr][lc])     = rB0;
        *(v8bf*)(&Bs[buf][lr][lc + 8]) = rB1;
        __syncthreads();

        // issue next tile's global loads (consumed at next iteration's store)
        if (it + 1 < nk) {
            const __bf16* ag2 = ag + (it + 1) * 32;
            const __bf16* bg2 = bg + (it + 1) * 32;
            rA0 = *(const v8bf*)(ag2);
            rA1 = *(const v8bf*)(ag2 + 8);
            rB0 = *(const v8bf*)(bg2);
            rB1 = *(const v8bf*)(bg2 + 8);
        }

        // per-lane fragments (ISA 7.12.2, 16-bit 16x32 layout):
        // elements 0..7 = K[half*8..], elements 8..15 = K[16+half*8..]
        v16bf a[2], b[4];
#pragma unroll
        for (int mt = 0; mt < 2; ++mt) {
            v8bf lo = *(const v8bf*)(&As[buf][wm * 32 + mt * 16 + l16][half * 8]);
            v8bf hi = *(const v8bf*)(&As[buf][wm * 32 + mt * 16 + l16][16 + half * 8]);
            a[mt] = __builtin_shufflevector(lo, hi, 0,1,2,3,4,5,6,7,8,9,10,11,12,13,14,15);
        }
#pragma unroll
        for (int nt = 0; nt < 4; ++nt) {
            v8bf lo = *(const v8bf*)(&Bs[buf][wn * 64 + nt * 16 + l16][half * 8]);
            v8bf hi = *(const v8bf*)(&Bs[buf][wn * 64 + nt * 16 + l16][16 + half * 8]);
            b[nt] = __builtin_shufflevector(lo, hi, 0,1,2,3,4,5,6,7,8,9,10,11,12,13,14,15);
        }
#pragma unroll
        for (int mt = 0; mt < 2; ++mt)
#pragma unroll
            for (int nt = 0; nt < 4; ++nt)
                acc[mt][nt] = __builtin_amdgcn_wmma_f32_16x16x32_bf16(
                    false, a[mt], false, b[nt], (short)0, acc[mt][nt], false, false);

        buf ^= 1;
    }

    // epilogue: D layout = VGPR r -> M = r + 8*half, N = lane%16
    float bv[4];
#pragma unroll
    for (int nt = 0; nt < 4; ++nt)
        bv[nt] = biasp[col0 + wn * 64 + nt * 16 + l16];

#pragma unroll
    for (int mt = 0; mt < 2; ++mt) {
#pragma unroll
        for (int nt = 0; nt < 4; ++nt) {
            const int colL = col0 + wn * 64 + nt * 16 + l16;
#pragma unroll
            for (int r = 0; r < 8; ++r) {
                const long rowL = row0 + wm * 32 + mt * 16 + half * 8 + r;
                float v = acc[mt][nt][r] + bv[nt];
                const long ci = rowL * (long)ldc + colL;
                if constexpr (EPI == EPI_STORE_F32) {
                    Cf[ci] = v;
                } else if constexpr (EPI == EPI_RELU_F32) {
                    Cf[ci] = fmaxf(v, 0.f);
                } else if constexpr (EPI == EPI_ACCUM_F32) {
                    Cf[ci] += v;
                } else if constexpr (EPI == EPI_RELU_BF16) {
                    Cb[ci] = (__bf16)fmaxf(v, 0.f);
                } else if constexpr (EPI == EPI_SILU_MUL_BF16) {
                    float g = aux1[rowL * (long)lda1 + colL];
                    float s = g / (1.f + __expf(-g));          // silu(g) * u
                    Cb[ci] = (__bf16)(s * v);
                } else { // EPI_RESID_F32: ffn delta + pre-FFN out + agent_x residual
                    Cf[ci] = v + aux1[rowL * (long)lda1 + colL] + aux2[rowL * (long)lda2 + colL];
                }
            }
        }
    }
}

// ---------------------------------------------------------------------------
// block-wide (256 threads) sum & sumsq
// ---------------------------------------------------------------------------
__device__ inline float2 blockSums(float v, float* sm)
{
    float a = v, b = v * v;
#pragma unroll
    for (int off = 16; off > 0; off >>= 1) {
        a += __shfl_xor(a, off, 32);
        b += __shfl_xor(b, off, 32);
    }
    const int lane = threadIdx.x & 31, wid = threadIdx.x >> 5;
    __syncthreads();
    if (lane == 0) { sm[wid] = a; sm[8 + wid] = b; }
    __syncthreads();
    float sa = 0.f, sb = 0.f;
#pragma unroll
    for (int i = 0; i < 8; ++i) { sa += sm[i]; sb += sm[8 + i]; }
    return make_float2(sa, sb);
}

// ---------------------------------------------------------------------------
// prep: gathers + all concat-LayerNorms -> bf16 GEMM inputs (1 block / agent)
// ---------------------------------------------------------------------------
__launch_bounds__(256)
__global__ void prep_kernel(const float* __restrict__ agent_x, const float* __restrict__ lane_x,
                            const float* __restrict__ poly_x,
                            const int* __restrict__ l2a_src, const int* __restrict__ g2a_src,
                            const int* __restrict__ other_src,
                            const float* __restrict__ l2a_ln_s, const float* __restrict__ l2a_ln_b,
                            const float* __restrict__ g2a_ln_s, const float* __restrict__ g2a_ln_b,
                            const float* __restrict__ oth_ln_s, const float* __restrict__ oth_ln_b,
                            __bf16* __restrict__ x_bf, __bf16* __restrict__ l2a_n,
                            __bf16* __restrict__ g2a_n, __bf16* __restrict__ oth_n)
{
    __shared__ float sm[16];
    const int n = blockIdx.x, j = threadIdx.x;

    const float xv = agent_x[(long)n * HD + j];
    const float lv = lane_x[(long)l2a_src[n] * HD + j];
    const float pv = poly_x[(long)g2a_src[n] * HD + j];
    float sv[3];
#pragma unroll
    for (int s = 0; s < 3; ++s)
        sv[s] = agent_x[(long)other_src[s * NTOT + n] * HD + j];

    const float2 Sx = blockSums(xv, sm);
    const float2 Sl = blockSums(lv, sm);
    const float2 Sp = blockSums(pv, sm);
    float2 Ss[3];
    for (int s = 0; s < 3; ++s) Ss[s] = blockSums(sv[s], sm);

    x_bf[(long)n * HD + j] = (__bf16)xv;

    {   // l2a: LN over cat([lane, x, lane]) -> 3H, lane counts twice in stats
        const float inv = 1.f / 768.f;
        const float mu = (2.f * Sl.x + Sx.x) * inv;
        const float m2 = (2.f * Sl.y + Sx.y) * inv;
        const float rs = rsqrtf(m2 - mu * mu + 1e-5f);
        __bf16* row = l2a_n + (long)n * 768;
        row[j]          = (__bf16)((lv - mu) * rs * l2a_ln_s[j]          + l2a_ln_b[j]);
        row[HD + j]     = (__bf16)((xv - mu) * rs * l2a_ln_s[HD + j]     + l2a_ln_b[HD + j]);
        row[2 * HD + j] = (__bf16)((lv - mu) * rs * l2a_ln_s[2 * HD + j] + l2a_ln_b[2 * HD + j]);
    }
    {   // g2a: LN over cat([poly, x]) -> 2H
        const float inv = 1.f / 512.f;
        const float mu = (Sp.x + Sx.x) * inv;
        const float m2 = (Sp.y + Sx.y) * inv;
        const float rs = rsqrtf(m2 - mu * mu + 1e-5f);
        __bf16* row = g2a_n + (long)n * 512;
        row[j]      = (__bf16)((pv - mu) * rs * g2a_ln_s[j]      + g2a_ln_b[j]);
        row[HD + j] = (__bf16)((xv - mu) * rs * g2a_ln_s[HD + j] + g2a_ln_b[HD + j]);
    }
    for (int s = 0; s < 3; ++s) {   // other: per-source-type LN over cat([src, x, src])
        const float inv = 1.f / 768.f;
        const float mu = (2.f * Ss[s].x + Sx.x) * inv;
        const float m2 = (2.f * Ss[s].y + Sx.y) * inv;
        const float rs = rsqrtf(m2 - mu * mu + 1e-5f);
        const float* ls = oth_ln_s + (long)s * 768;
        const float* lb = oth_ln_b + (long)s * 768;
        __bf16* row = oth_n + (long)s * NTOT * 768 + (long)n * 768;
        const float q = sv[s];
        row[j]          = (__bf16)((q  - mu) * rs * ls[j]          + lb[j]);
        row[HD + j]     = (__bf16)((xv - mu) * rs * ls[HD + j]     + lb[HD + j]);
        row[2 * HD + j] = (__bf16)((q  - mu) * rs * ls[2 * HD + j] + lb[2 * HD + j]);
    }
}

// FFN pre-norm: y = LN(out) * ffn_ln_s[t] + ffn_ln_b[t], bf16 (1 block / agent)
__launch_bounds__(256)
__global__ void ln_ffn_kernel(const float* __restrict__ outf,
                              const float* __restrict__ ln_s, const float* __restrict__ ln_b,
                              __bf16* __restrict__ y)
{
    __shared__ float sm[16];
    const int n = blockIdx.x, j = threadIdx.x, t = n / NPT;
    const float v = outf[(long)n * HD + j];
    const float2 S = blockSums(v, sm);
    const float mu = S.x * (1.f / 256.f);
    const float m2 = S.y * (1.f / 256.f);
    const float rs = rsqrtf(m2 - mu * mu + 1e-5f);
    y[(long)n * HD + j] = (__bf16)((v - mu) * rs * ln_s[t * HD + j] + ln_b[t * HD + j]);
}

// fp32 [batch][K][N] -> bf16 [batch][N][K] (N-major so B tiles load like A tiles)
__global__ void wconv_kernel(const float* __restrict__ W, __bf16* __restrict__ Wt,
                             int K, int Nc, int batch)
{
    const long total = (long)batch * K * Nc;
    for (long i = (long)blockIdx.x * blockDim.x + threadIdx.x; i < total;
         i += (long)gridDim.x * blockDim.x) {
        const long b = i / ((long)K * Nc);
        const long rem = i - b * (long)K * Nc;
        const long k = rem / Nc, nn = rem - k * Nc;
        Wt[b * (long)K * Nc + nn * (long)K + k] = (__bf16)W[i];
    }
}

__global__ void cvt_f32_bf16_kernel(const float* __restrict__ src, __bf16* __restrict__ dst, long count)
{
    for (long i = (long)blockIdx.x * blockDim.x + threadIdx.x; i < count;
         i += (long)gridDim.x * blockDim.x)
        dst[i] = (__bf16)src[i];
}

// ---------------------------------------------------------------------------
extern "C" void kernel_launch(void* const* d_in, const int* in_sizes, int n_in,
                              void* d_out, int out_size, void* d_ws, size_t ws_size,
                              hipStream_t stream)
{
    (void)in_sizes; (void)n_in; (void)out_size; (void)ws_size;
    const float* agent_x  = (const float*)d_in[0];
    const float* lane_x   = (const float*)d_in[1];
    const float* poly_x   = (const float*)d_in[2];
    const int*   l2a_src  = (const int*)d_in[3];
    const int*   g2a_src  = (const int*)d_in[4];
    const int*   other_src= (const int*)d_in[5];
    const float* self_W   = (const float*)d_in[6];
    const float* self_b   = (const float*)d_in[7];
    const float* out_W    = (const float*)d_in[8];
    const float* out_b    = (const float*)d_in[9];
    const float* ffn_ln_s = (const float*)d_in[10];
    const float* ffn_ln_b = (const float*)d_in[11];
    const float* ffn_w1   = (const float*)d_in[12];
    const float* ffn_b1   = (const float*)d_in[13];
    const float* ffn_w2   = (const float*)d_in[14];
    const float* ffn_b2   = (const float*)d_in[15];
    const float* ffn_w3   = (const float*)d_in[16];
    const float* ffn_b3   = (const float*)d_in[17];
    const float* l2a_ln_s = (const float*)d_in[18];
    const float* l2a_ln_b = (const float*)d_in[19];
    const float* l2a_w1   = (const float*)d_in[20];
    const float* l2a_b1   = (const float*)d_in[21];
    const float* l2a_w2   = (const float*)d_in[22];
    const float* l2a_b2   = (const float*)d_in[23];
    const float* g2a_ln_s = (const float*)d_in[24];
    const float* g2a_ln_b = (const float*)d_in[25];
    const float* g2a_w1   = (const float*)d_in[26];
    const float* g2a_b1   = (const float*)d_in[27];
    const float* g2a_w2   = (const float*)d_in[28];
    const float* g2a_b2   = (const float*)d_in[29];
    const float* oth_ln_s = (const float*)d_in[30];
    const float* oth_ln_b = (const float*)d_in[31];
    const float* oth_w1   = (const float*)d_in[32];
    const float* oth_b1   = (const float*)d_in[33];
    const float* oth_w2   = (const float*)d_in[34];
    const float* oth_b2   = (const float*)d_in[35];

    char* ws = (char*)d_ws;
    size_t o = 0;
    auto alloc = [&](size_t bytes) { size_t r = (o + 255) & ~(size_t)255; o = r + bytes; return r; };

    // bf16 weights, N-major per matrix ([Ncols][K]); per-type slabs contiguous
    const size_t o_selfWt = alloc((size_t)3 * 256 * 256 * 2);
    const size_t o_outWt  = alloc((size_t)3 * 256 * 1024 * 2);
    const size_t o_fw1t   = alloc((size_t)3 * 1024 * 256 * 2);
    const size_t o_fw2t   = alloc((size_t)3 * 256 * 1024 * 2);
    const size_t o_fw3t   = alloc((size_t)3 * 1024 * 256 * 2);
    const size_t o_l2aw1t = alloc((size_t)1024 * 768 * 2);
    const size_t o_l2aw2t = alloc((size_t)256 * 1024 * 2);
    const size_t o_g2aw1t = alloc((size_t)1024 * 512 * 2);
    const size_t o_g2aw2t = alloc((size_t)256 * 1024 * 2);
    const size_t o_othw1t = alloc((size_t)3 * 1024 * 768 * 2);
    const size_t o_othw2t = alloc((size_t)3 * 256 * 1024 * 2);
    // activations (with overlays)
    const size_t o_xbf    = alloc((size_t)NTOT * 256 * 2);
    const size_t o_l2an   = alloc((size_t)NTOT * 768 * 2);   // reused later as fused_bf16
    const size_t o_g2an   = alloc((size_t)NTOT * 512 * 2);   // (l2an+g2an region = 62.9MB >= 50.3MB)
    const size_t o_othn   = alloc((size_t)3 * NTOT * 768 * 2); // reused later as g_f32 (113MB >= 100.7MB)
    const size_t o_hbuf   = alloc((size_t)NTOT * 1024 * 2);
    const size_t o_fused  = alloc((size_t)NTOT * 1024 * 4);  // reused later as gu_bf16
    const size_t o_outf   = alloc((size_t)NTOT * 256 * 4);
    const size_t o_ybf    = alloc((size_t)NTOT * 256 * 2);

    __bf16* selfWt = (__bf16*)(ws + o_selfWt);
    __bf16* outWt  = (__bf16*)(ws + o_outWt);
    __bf16* fw1t   = (__bf16*)(ws + o_fw1t);
    __bf16* fw2t   = (__bf16*)(ws + o_fw2t);
    __bf16* fw3t   = (__bf16*)(ws + o_fw3t);
    __bf16* l2aw1t = (__bf16*)(ws + o_l2aw1t);
    __bf16* l2aw2t = (__bf16*)(ws + o_l2aw2t);
    __bf16* g2aw1t = (__bf16*)(ws + o_g2aw1t);
    __bf16* g2aw2t = (__bf16*)(ws + o_g2aw2t);
    __bf16* othw1t = (__bf16*)(ws + o_othw1t);
    __bf16* othw2t = (__bf16*)(ws + o_othw2t);
    __bf16* x_bf   = (__bf16*)(ws + o_xbf);
    __bf16* l2an   = (__bf16*)(ws + o_l2an);
    __bf16* g2an   = (__bf16*)(ws + o_g2an);
    __bf16* othn   = (__bf16*)(ws + o_othn);
    __bf16* hbuf   = (__bf16*)(ws + o_hbuf);
    float*  fused  = (float*)(ws + o_fused);
    float*  outf   = (float*)(ws + o_outf);
    __bf16* ybf    = (__bf16*)(ws + o_ybf);
    __bf16* fusedbf= (__bf16*)(ws + o_l2an);   // overlay (l2a/g2a norms dead)
    float*  gbuf   = (float*)(ws + o_othn);    // overlay (oth norms dead)
    __bf16* gubf   = (__bf16*)(ws + o_fused);  // overlay (fused_f32 dead)
    float*  fin    = (float*)d_out;

    // --- weight convert/transpose to bf16 N-major ---
    wconv_kernel<<<1024, 256, 0, stream>>>(self_W, selfWt,  256,  256, 3);
    wconv_kernel<<<1024, 256, 0, stream>>>(out_W,  outWt,  1024,  256, 3);
    wconv_kernel<<<1024, 256, 0, stream>>>(ffn_w1, fw1t,    256, 1024, 3);
    wconv_kernel<<<1024, 256, 0, stream>>>(ffn_w2, fw2t,   1024,  256, 3);
    wconv_kernel<<<1024, 256, 0, stream>>>(ffn_w3, fw3t,    256, 1024, 3);
    wconv_kernel<<<1024, 256, 0, stream>>>(l2a_w1, l2aw1t,  768, 1024, 1);
    wconv_kernel<<<1024, 256, 0, stream>>>(l2a_w2, l2aw2t, 1024,  256, 1);
    wconv_kernel<<<1024, 256, 0, stream>>>(g2a_w1, g2aw1t,  512, 1024, 1);
    wconv_kernel<<<1024, 256, 0, stream>>>(g2a_w2, g2aw2t, 1024,  256, 1);
    wconv_kernel<<<1024, 256, 0, stream>>>(oth_w1, othw1t,  768, 1024, 3);
    wconv_kernel<<<1024, 256, 0, stream>>>(oth_w2, othw2t, 1024,  256, 3);

    // --- gathers + LayerNorms -> bf16 inputs ---
    prep_kernel<<<NTOT, 256, 0, stream>>>(agent_x, lane_x, poly_x, l2a_src, g2a_src, other_src,
                                          l2a_ln_s, l2a_ln_b, g2a_ln_s, g2a_ln_b, oth_ln_s, oth_ln_b,
                                          x_bf, l2an, g2an, othn);

    const dim3 blk(256);
    const dim3 g256(256 / 128, NTOT / 128);    // Ncols=256  -> 2 x 192
    const dim3 g1024(1024 / 128, NTOT / 128);  // Ncols=1024 -> 8 x 192

    // self branch: relu(x @ self_W[t] + self_b[t]) -> fused[:, 0:256]
    gemm_bf16_wmma<EPI_RELU_F32><<<g256, blk, 0, stream>>>(
        x_bf, 256, selfWt, 256L * 256, 256, self_b, 256,
        fused + 0, nullptr, 1024, nullptr, 0, nullptr, 0, 256);

    // l2a MLP (shared weights) -> fused[:, 256:512]
    gemm_bf16_wmma<EPI_RELU_BF16><<<g1024, blk, 0, stream>>>(
        l2an, 768, l2aw1t, 0, 768, l2a_b1, 0,
        nullptr, hbuf, 1024, nullptr, 0, nullptr, 0, 1024);
    gemm_bf16_wmma<EPI_STORE_F32><<<g256, blk, 0, stream>>>(
        hbuf, 1024, l2aw2t, 0, 1024, l2a_b2, 0,
        fused + 256, nullptr, 1024, nullptr, 0, nullptr, 0, 256);

    // g2a MLP (shared weights) -> fused[:, 512:768]
    gemm_bf16_wmma<EPI_RELU_BF16><<<g1024, blk, 0, stream>>>(
        g2an, 512, g2aw1t, 0, 512, g2a_b1, 0,
        nullptr, hbuf, 1024, nullptr, 0, nullptr, 0, 1024);
    gemm_bf16_wmma<EPI_STORE_F32><<<g256, blk, 0, stream>>>(
        hbuf, 1024, g2aw2t, 0, 1024, g2a_b2, 0,
        fused + 512, nullptr, 1024, nullptr, 0, nullptr, 0, 256);

    // other branch: per source type s, MLP then accumulate -> fused[:, 768:1024]
    for (int s = 0; s < 3; ++s) {
        gemm_bf16_wmma<EPI_RELU_BF16><<<g1024, blk, 0, stream>>>(
            othn + (long)s * NTOT * 768, 768, othw1t + (long)s * 1024 * 768, 0, 768,
            oth_b1 + (long)s * 1024, 0,
            nullptr, hbuf, 1024, nullptr, 0, nullptr, 0, 1024);
        if (s == 0)
            gemm_bf16_wmma<EPI_STORE_F32><<<g256, blk, 0, stream>>>(
                hbuf, 1024, othw2t + (long)s * 256 * 1024, 0, 1024, oth_b2 + (long)s * 256, 0,
                fused + 768, nullptr, 1024, nullptr, 0, nullptr, 0, 256);
        else
            gemm_bf16_wmma<EPI_ACCUM_F32><<<g256, blk, 0, stream>>>(
                hbuf, 1024, othw2t + (long)s * 256 * 1024, 0, 1024, oth_b2 + (long)s * 256, 0,
                fused + 768, nullptr, 1024, nullptr, 0, nullptr, 0, 256);
    }

    // fused f32 -> bf16
    cvt_f32_bf16_kernel<<<4096, 256, 0, stream>>>(fused, fusedbf, (long)NTOT * 1024);

    // out_fc: out = fused @ out_W[t] + out_b[t]
    gemm_bf16_wmma<EPI_STORE_F32><<<g256, blk, 0, stream>>>(
        fusedbf, 1024, outWt, 256L * 1024, 1024, out_b, 256,
        outf, nullptr, 256, nullptr, 0, nullptr, 0, 256);

    // FFN: y = LN(out); g = y@w1+b1; gu = silu(g)*(y@w3+b3); final = (gu@w2+b2) + out + x
    ln_ffn_kernel<<<NTOT, 256, 0, stream>>>(outf, ffn_ln_s, ffn_ln_b, ybf);
    gemm_bf16_wmma<EPI_STORE_F32><<<g1024, blk, 0, stream>>>(
        ybf, 256, fw1t, 1024L * 256, 256, ffn_b1, 1024,
        gbuf, nullptr, 1024, nullptr, 0, nullptr, 0, 1024);
    gemm_bf16_wmma<EPI_SILU_MUL_BF16><<<g1024, blk, 0, stream>>>(
        ybf, 256, fw3t, 1024L * 256, 256, ffn_b3, 1024,
        nullptr, gubf, 1024, gbuf, 1024, nullptr, 0, 1024);
    gemm_bf16_wmma<EPI_RESID_F32><<<g256, blk, 0, stream>>>(
        gubf, 1024, fw2t, 256L * 1024, 1024, ffn_b2, 256,
        fin, nullptr, 256, outf, 256, agent_x, 256, 256);
}
